// BuildModel_74096775790700
// MI455X (gfx1250) — matrix operand order repaired and checked by reading, not compiled, not run on hardware
//
#include <hip/hip_runtime.h>

// ---------------------------------------------------------------------------
// NTM (CNN + LSTM controller + memory heads) for MI455X / gfx1250, wave32.
// GEMMs use v_wmma_f32_16x16x32_f16; weights pre-swizzled once into per-lane
// WMMA fragment order. Per-timestep dependency expressed via stream ordering.
// 4 kernels per timestep: gemm_z -> lstm_pointwise -> gemm_p -> heads_update.
// ---------------------------------------------------------------------------

typedef _Float16 half_t;
typedef __attribute__((ext_vector_type(16))) _Float16 v16h;
typedef __attribute__((ext_vector_type(8)))  _Float16 v8h;
typedef __attribute__((ext_vector_type(8)))  float    v8f;

#define DEVFN static __device__ __forceinline__

DEVFN float sigm(float x)      { return 1.f / (1.f + expf(-x)); }
DEVFN float softplusf(float x) { return (x > 15.f) ? x : log1pf(expf(x)); }
DEVFN float clip20(float x)    { return fminf(fmaxf(x, -20.f), 20.f); }

// ---------------- CNN front-end (direct conv, run once) --------------------

// conv 3x3 (1->16) + relu + 2x2 maxpool : in [256,64,64] -> out [256,32,32,16]
__global__ void conv1_pool(const float* __restrict__ in, const float* __restrict__ w,
                           const float* __restrict__ bias, float* __restrict__ out) {
  int t = blockIdx.x * 256 + threadIdx.x;          // 4,194,304 threads
  int c = t & 15, x = (t >> 4) & 31, y = (t >> 9) & 31, b = t >> 14;
  float best = -1e30f;
  #pragma unroll
  for (int py = 0; py < 2; ++py)
  #pragma unroll
  for (int px = 0; px < 2; ++px) {
    int oy = 2 * y + py, ox = 2 * x + px;
    float acc = bias[c];
    #pragma unroll
    for (int ky = 0; ky < 3; ++ky) {
      int iy = oy + ky - 1; if ((unsigned)iy >= 64u) continue;
      #pragma unroll
      for (int kx = 0; kx < 3; ++kx) {
        int ix = ox + kx - 1; if ((unsigned)ix >= 64u) continue;
        acc += in[((size_t)b * 64 + iy) * 64 + ix] * w[(ky * 3 + kx) * 16 + c];
      }
    }
    best = fmaxf(best, fmaxf(acc, 0.f));
  }
  out[t] = best;   // layout ((b*32+y)*32+x)*16+c == t
}

// conv 3x3 (16->32) + relu + maxpool : -> xseq f16 [T=256][B=256][32]
__global__ void conv2_pool(const float* __restrict__ in, const float* __restrict__ w,
                           const float* __restrict__ bias, half_t* __restrict__ xseq) {
  int t = blockIdx.x * 256 + threadIdx.x;          // 2,097,152 threads
  int c = t & 31, x = (t >> 5) & 15, y = (t >> 9) & 15, b = t >> 13;
  float best = -1e30f;
  for (int py = 0; py < 2; ++py)
  for (int px = 0; px < 2; ++px) {
    int oy = 2 * y + py, ox = 2 * x + px;
    float acc = bias[c];
    for (int ky = 0; ky < 3; ++ky) {
      int iy = oy + ky - 1; if ((unsigned)iy >= 32u) continue;
      for (int kx = 0; kx < 3; ++kx) {
        int ix = ox + kx - 1; if ((unsigned)ix >= 32u) continue;
        const float* ip = in + (((size_t)b * 32 + iy) * 32 + ix) * 16;
        const float* wp = w + ((ky * 3 + kx) * 16) * 32 + c;
        #pragma unroll
        for (int ci = 0; ci < 16; ++ci) acc += ip[ci] * wp[ci * 32];
      }
    }
    best = fmaxf(best, fmaxf(acc, 0.f));
  }
  int tt = y * 16 + x;
  xseq[((size_t)tt * 256 + b) * 32 + c] = (half_t)best;
}

// ------------- weight pre-swizzle into WMMA B-fragment order ---------------
// B fragment (32x16 f16): lane&15 = n; lanes 0-15 hold K 0..15, lanes 16-31
// hold K 16..31; half i of v16h <-> K = base+i. Stored as
// WF[((nt*KT + kt)*32 + lane)*16 + i]  -> one contiguous 32B load per lane.

// combined LSTM weights, K padded 308->320: rows 0..51 = lstm_wx, 52..63 = 0,
// 64..319 = lstm_wh. N = 1024, KT = 10, NT = 64.
__global__ void prep_wz(const float* __restrict__ wx, const float* __restrict__ wh,
                        half_t* __restrict__ WF) {
  int t = blockIdx.x * 256 + threadIdx.x;          // 327,680 threads
  int i = t & 15, lane = (t >> 4) & 31, rest = t >> 9;
  int kt = rest % 10, nt = rest / 10;
  int n = nt * 16 + (lane & 15);
  int k = kt * 32 + ((lane & 16) ? 16 : 0) + i;
  float v = 0.f;
  if (k < 52)       v = wx[(size_t)k * 1024 + n];
  else if (k >= 64) v = wh[(size_t)(k - 64) * 1024 + n];
  WF[t] = (half_t)v;
}

// head weights, N padded 92->96. K = 256 (KT=8), NT = 6.
__global__ void prep_wp(const float* __restrict__ hw, half_t* __restrict__ WF) {
  int t = blockIdx.x * 256 + threadIdx.x;          // 24,576 threads
  int i = t & 15, lane = (t >> 4) & 31, rest = t >> 9;
  int kt = rest & 7, nt = rest >> 3;
  int n = nt * 16 + (lane & 15);
  int k = kt * 32 + ((lane & 16) ? 16 : 0) + i;
  WF[t] = (half_t)((n < 92) ? hw[(size_t)k * 92 + n] : 0.f);
}

// ------------------------- state initialization ----------------------------

__global__ void init_state(float* __restrict__ Mmem, float* __restrict__ wprev,
                           float* __restrict__ hbuf, float* __restrict__ cbuf,
                           half_t* __restrict__ h16, float* __restrict__ reads,
                           half_t* __restrict__ reads16) {
  int t = blockIdx.x * 256 + threadIdx.x;          // 655,360 threads
  Mmem[t] = 1e-6f;
  if (t < 65536) { wprev[t] = 1.f / 128.f; hbuf[t] = 0.f; cbuf[t] = 0.f; h16[t] = (half_t)0.f; }
  if (t < 8192)  reads16[t] = ((t & 31) < 20) ? (half_t)1e-6f : (half_t)0.f;
  if (t < 5120)  reads[t] = 1e-6f;
}

// ----------------------- WMMA fragment loaders -----------------------------
// A fragment (16x32 f16, row-major source, stride lda):
// lanes 0-15: m=lane, K 0..15 ; lanes 16-31: m=lane-16, K 16..31.
DEVFN v16h load_a_frag(const half_t* __restrict__ A, int lda, int m0, int k0, int lane) {
  int m  = m0 + (lane & 15);
  int kb = k0 + ((lane & 16) ? 16 : 0);
  const v8h* p = (const v8h*)(A + (size_t)m * lda + kb);
  v8h lo = p[0], hi = p[1];                        // K kb..kb+7, kb+8..kb+15
  v16h a;
  #pragma unroll
  for (int i = 0; i < 8; ++i) { a[i] = lo[i]; a[8 + i] = hi[i]; }
  return a;
}

// --------------------- per-step GEMM: z = ci @ Wz --------------------------
// z [256 x 1024], A = [ x_t (32) | reads (20, padded 32) | h (256) ], K=320.
// Each wave computes two adjacent N-tiles sharing its A fragments (A reuse,
// 2 WMMAs per A-load clause). 512 waves -> grid 64 x 256 threads.
__global__ void gemm_z(const half_t* __restrict__ xt, const half_t* __restrict__ reads16,
                       const half_t* __restrict__ h16, const half_t* __restrict__ WF,
                       float* __restrict__ z) {
  int wid  = (blockIdx.x * 256 + threadIdx.x) >> 5;   // 0..511
  int lane = threadIdx.x & 31;
  int mt = wid & 15, ntp = wid >> 4;                  // ntp 0..31
  int lrow = lane & 15, hi = (lane >> 4) & 1;
  v8f acc0 = {}, acc1 = {};
  #pragma unroll
  for (int kt = 0; kt < 10; ++kt) {
    v16h a;
    if      (kt == 0) a = load_a_frag(xt,      32,  mt * 16, 0,             lane);
    else if (kt == 1) a = load_a_frag(reads16, 32,  mt * 16, 0,             lane);
    else              a = load_a_frag(h16,     256, mt * 16, (kt - 2) * 32, lane);
    v16h b0 = *(const v16h*)(WF + ((((size_t)ntp * 2 + 0) * 10 + kt) * 32 + lane) * 16);
    v16h b1 = *(const v16h*)(WF + ((((size_t)ntp * 2 + 1) * 10 + kt) * 32 + lane) * 16);
    acc0 = __builtin_amdgcn_wmma_f32_16x16x32_f16(false, a, false, b0, (short)0, acc0,
                                                  false, false);
    acc1 = __builtin_amdgcn_wmma_f32_16x16x32_f16(false, a, false, b1, (short)0, acc1,
                                                  false, false);
  }
  int mbase = mt * 16 + hi * 8;
  int n0 = (ntp * 2) * 16 + lrow, n1 = n0 + 16;
  #pragma unroll
  for (int r = 0; r < 8; ++r) {
    z[(size_t)(mbase + r) * 1024 + n0] = acc0[r];
    z[(size_t)(mbase + r) * 1024 + n1] = acc1[r];
  }
}

// -------------------------- LSTM pointwise ---------------------------------
__global__ void lstm_pointwise(const float* __restrict__ z, const float* __restrict__ lb,
                               float* __restrict__ cbuf, float* __restrict__ hbuf,
                               half_t* __restrict__ h16) {
  int t = blockIdx.x * 256 + threadIdx.x;          // 65,536 threads
  int b = t >> 8, u = t & 255;
  const float* zr = z + (size_t)b * 1024;
  float zi = zr[u]       + lb[u];
  float zf = zr[256 + u] + lb[256 + u];
  float zg = zr[512 + u] + lb[512 + u];
  float zo = zr[768 + u] + lb[768 + u];
  float c = sigm(zf) * cbuf[t] + sigm(zi) * tanhf(zg);
  float h = sigm(zo) * tanhf(c);
  cbuf[t] = c; hbuf[t] = h; h16[t] = (half_t)h;
}

// -------------------- head-param GEMM: p = clip(h @ Wp + b) ----------------
// 96 tiles (M=16 x N=6). Grid 12 x 256.
__global__ void gemm_p(const half_t* __restrict__ h16, const half_t* __restrict__ WF,
                       const float* __restrict__ hb, float* __restrict__ pbuf) {
  int wid  = (blockIdx.x * 256 + threadIdx.x) >> 5;   // 0..95
  int lane = threadIdx.x & 31;
  int mt = wid & 15, nt = wid >> 4;
  int lrow = lane & 15, hi = (lane >> 4) & 1;
  v8f acc = {};
  #pragma unroll
  for (int kt = 0; kt < 8; ++kt) {
    v16h a  = load_a_frag(h16, 256, mt * 16, kt * 32, lane);
    v16h bf = *(const v16h*)(WF + (((size_t)nt * 8 + kt) * 32 + lane) * 16);
    acc = __builtin_amdgcn_wmma_f32_16x16x32_f16(false, a, false, bf, (short)0, acc,
                                                 false, false);
  }
  int n = nt * 16 + lrow;
  if (n < 92) {
    float bias = hb[n];
    int mbase = mt * 16 + hi * 8;
    #pragma unroll
    for (int r = 0; r < 8; ++r)
      pbuf[(size_t)(mbase + r) * 96 + n] = clip20(acc[r] + bias);
  }
}

// ------- fused heads: addressing (both heads) + read + erase/add write -----
// One block per batch element b: 256 blocks x 128 threads (thread = slot n).
// Previous M[b] is snapshotted in LDS, so the read vector and the erase/add
// update both see pre-write memory (matching the reference semantics).
__global__ void heads_update(const float* __restrict__ pbuf, float* __restrict__ Mmem,
                             float* __restrict__ wprev, float* __restrict__ reads,
                             half_t* __restrict__ reads16) {
  int b = blockIdx.x;                              // 0..255
  int n = threadIdx.x;                             // 0..127
  __shared__ float Ml[128 * 20];
  __shared__ float red[128];
  __shared__ float wg[128];
  __shared__ float wh[2][128];
  __shared__ float eL[20], aL[20];
  const float EPSc = 1e-8f;

  float mr[20];
  float* Mrow = Mmem + ((size_t)b * 128 + n) * 20;
  float mn2 = 0.f;
  #pragma unroll
  for (int j = 0; j < 20; ++j) { mr[j] = Mrow[j]; Ml[n * 20 + j] = mr[j]; mn2 += mr[j] * mr[j]; }
  float minv = 1.f / (sqrtf(mn2) + EPSc);

  if (n < 20) {                                    // erase/add activations
    eL[n] = sigm(pbuf[(size_t)b * 96 + 52 + n]);
    aL[n] = tanhf(pbuf[(size_t)b * 96 + 72 + n]);
  }

  for (int head = 0; head < 2; ++head) {
    const float* pp = pbuf + (size_t)b * 96 + head * 26;
    float tk[20], kn2 = 0.f, dot = 0.f;
    #pragma unroll
    for (int j = 0; j < 20; ++j) { tk[j] = tanhf(pp[j]); kn2 += tk[j] * tk[j]; dot += tk[j] * mr[j]; }
    float kinv  = 1.f / (sqrtf(kn2) + EPSc);
    float beta  = softplusf(pp[20]);
    float gate  = sigm(pp[21]);
    float a0 = pp[22], a1 = pp[23], a2 = pp[24];
    float smx = fmaxf(a0, fmaxf(a1, a2));
    float e0 = expf(a0 - smx), e1 = expf(a1 - smx), e2 = expf(a2 - smx);
    float einv = 1.f / (e0 + e1 + e2);
    float s0 = e0 * einv, s1 = e1 * einv, s2 = e2 * einv;
    float gamma = softplusf(pp[25]) + 1.f;
    float logit = beta * dot * kinv * minv;

    // content softmax over n
    red[n] = logit; __syncthreads();
    for (int st = 64; st > 0; st >>= 1) { if (n < st) red[n] = fmaxf(red[n], red[n + st]); __syncthreads(); }
    float mx = red[0]; __syncthreads();
    float ex = expf(logit - mx);
    red[n] = ex; __syncthreads();
    for (int st = 64; st > 0; st >>= 1) { if (n < st) red[n] += red[n + st]; __syncthreads(); }
    float wc = ex / red[0]; __syncthreads();

    // interpolate, circular shift, sharpen, renormalize
    float wgn = gate * wc + (1.f - gate) * wprev[((size_t)b * 2 + head) * 128 + n];
    wg[n] = wgn; __syncthreads();
    float wt = s0 * wg[(n + 1) & 127] + s1 * wgn + s2 * wg[(n + 127) & 127];
    float wp = powf(wt + EPSc, gamma);
    red[n] = wp; __syncthreads();
    for (int st = 64; st > 0; st >>= 1) { if (n < st) red[n] += red[n + st]; __syncthreads(); }
    float w = wp / red[0];
    wprev[((size_t)b * 2 + head) * 128 + n] = w;
    wh[head][n] = w;
    __syncthreads();
  }

  // read vector (head 0) from previous memory snapshot
  if (n < 20) {
    float r = 0.f;
    for (int q = 0; q < 128; ++q) r += wh[0][q] * Ml[q * 20 + n];
    reads[b * 20 + n] = r;
    reads16[b * 32 + n] = (half_t)r;
  }
  // erase/add (write head = index 1) using snapshotted previous M
  float ww = wh[1][n];
  #pragma unroll
  for (int j = 0; j < 20; ++j)
    Mrow[j] = mr[j] * (1.f - ww * eL[j]) + ww * aL[j];
}

// --------------- final: out head + dense + softmax -> d_out ----------------
__global__ void final_out(const float* __restrict__ hbuf, const float* __restrict__ reads,
                          const float* __restrict__ ow, const float* __restrict__ ob,
                          const float* __restrict__ dw, const float* __restrict__ db,
                          float* __restrict__ out) {
  int b = threadIdx.x;                             // 256 threads, 1 block
  float o[8];
  #pragma unroll
  for (int oc = 0; oc < 8; ++oc) o[oc] = ob[oc];
  for (int i = 0; i < 256; ++i) {
    float hv = hbuf[b * 256 + i];
    #pragma unroll
    for (int oc = 0; oc < 8; ++oc) o[oc] += hv * ow[i * 8 + oc];
  }
  for (int j = 0; j < 20; ++j) {
    float rv = reads[b * 20 + j];
    #pragma unroll
    for (int oc = 0; oc < 8; ++oc) o[oc] += rv * ow[(256 + j) * 8 + oc];
  }
  float l0 = db[0], l1 = db[1];
  #pragma unroll
  for (int oc = 0; oc < 8; ++oc) {
    float v = clip20(o[oc]);
    l0 += v * dw[oc * 2]; l1 += v * dw[oc * 2 + 1];
  }
  float m = fmaxf(l0, l1);
  float e0 = expf(l0 - m), e1 = expf(l1 - m), s = 1.f / (e0 + e1);
  out[b * 2] = e0 * s; out[b * 2 + 1] = e1 * s;
}

// ---------------------------------------------------------------------------

extern "C" void kernel_launch(void* const* d_in, const int* in_sizes, int n_in,
                              void* d_out, int out_size, void* d_ws, size_t ws_size,
                              hipStream_t stream) {
  const float* inputs  = (const float*)d_in[0];
  const float* conv1_w = (const float*)d_in[1];
  const float* conv1_b = (const float*)d_in[2];
  const float* conv2_w = (const float*)d_in[3];
  const float* conv2_b = (const float*)d_in[4];
  const float* lstm_wx = (const float*)d_in[5];
  const float* lstm_wh = (const float*)d_in[6];
  const float* lstm_b  = (const float*)d_in[7];
  const float* head_w  = (const float*)d_in[8];
  const float* head_b  = (const float*)d_in[9];
  const float* out_w   = (const float*)d_in[10];
  const float* out_b   = (const float*)d_in[11];
  const float* dense_w = (const float*)d_in[12];
  const float* dense_b = (const float*)d_in[13];
  float* out = (float*)d_out;

  char* base = (char*)d_ws;
  size_t off = 0;
  auto carve = [&](size_t bytes) -> void* {
    void* p = base + off;
    off = (off + bytes + 255) & ~(size_t)255;
    return p;
  };
  float*  c1out   = (float*) carve((size_t)4194304 * 4);  // [256,32,32,16]
  half_t* xseq    = (half_t*)carve((size_t)2097152 * 2);  // [T=256][B=256][32]
  half_t* WzF     = (half_t*)carve((size_t)327680  * 2);  // LSTM W frag
  half_t* WpF     = (half_t*)carve((size_t)24576   * 2);  // head W frag
  float*  z       = (float*) carve((size_t)262144  * 4);  // [256,1024]
  float*  hbuf    = (float*) carve((size_t)65536   * 4);
  half_t* h16     = (half_t*)carve((size_t)65536   * 2);
  float*  cbuf    = (float*) carve((size_t)65536   * 4);
  float*  pbuf    = (float*) carve((size_t)24576   * 4);  // [256,96]
  float*  wprev   = (float*) carve((size_t)65536   * 4);  // [256,2,128]
  float*  Mmem    = (float*) carve((size_t)655360  * 4);  // [256,128,20]
  float*  reads   = (float*) carve((size_t)5120    * 4);  // [256,20]
  half_t* reads16 = (half_t*)carve((size_t)8192    * 2);  // [256,32] (padded)
  if (off > ws_size) return;  // workspace too small; refuse to corrupt memory

  // front-end + one-time weight swizzle + state init
  conv1_pool<<<16384, 256, 0, stream>>>(inputs, conv1_w, conv1_b, c1out);
  conv2_pool<<<8192,  256, 0, stream>>>(c1out, conv2_w, conv2_b, xseq);
  prep_wz<<<1280, 256, 0, stream>>>(lstm_wx, lstm_wh, WzF);
  prep_wp<<<96,   256, 0, stream>>>(head_w, WpF);
  init_state<<<2560, 256, 0, stream>>>(Mmem, wprev, hbuf, cbuf, h16, reads, reads16);

  // 256 sequential NTM timesteps (dependency via stream ordering)
  for (int t = 0; t < 256; ++t) {
    const half_t* xt = xseq + (size_t)t * 256 * 32;
    gemm_z<<<64, 256, 0, stream>>>(xt, reads16, h16, WzF, z);
    lstm_pointwise<<<256, 256, 0, stream>>>(z, lstm_b, cbuf, hbuf, h16);
    gemm_p<<<12, 256, 0, stream>>>(h16, WpF, head_b, pbuf);
    heads_update<<<256, 128, 0, stream>>>(pbuf, Mmem, wprev, reads, reads16);
  }

  final_out<<<1, 256, 0, stream>>>(hbuf, reads, out_w, out_b, dense_w, dense_b, out);
}
